// RegionEncoder_14955076125387
// MI455X (gfx1250) — matrix operand order repaired and checked by reading, not compiled, not run on hardware
//
#include <hip/hip_runtime.h>
#include <hip/hip_bf16.h>

typedef __attribute__((ext_vector_type(16))) _Float16 v16h;
typedef __attribute__((ext_vector_type(8)))  _Float16 v8h;
typedef __attribute__((ext_vector_type(8)))  float    v8f;

#define BATCH 8
#define NPTS  12544

__device__ __forceinline__ float gelu_exact(float y) {
    return 0.5f * y * (1.0f + erff(y * 0.70710678118654752f));
}

// ---------------------------------------------------------------------------
// conv1 7x7 stride 7 : [B,4,448,448] -> [B,64,64,256] NHWC f16, LN + GELU fused
// one block per output pixel, 256 threads = 256 output channels
// ---------------------------------------------------------------------------
__global__ void k_conv1_ln_gelu(const float* __restrict__ img,
                                const float* __restrict__ msk,
                                const float* __restrict__ w,     // [256,4,7,7]
                                const float* __restrict__ bias,  // [256]
                                const float* __restrict__ lnw,
                                const float* __restrict__ lnb,
                                _Float16* __restrict__ y1) {
    int pix = blockIdx.x;                 // 0..32767  = b*4096 + oy*64 + ox
    int b = pix >> 12;
    int rem = pix & 4095;
    int oy = rem >> 6, ox = rem & 63;
    int t = threadIdx.x;                  // 256

    __shared__ float patch[196];
    __shared__ float sb[256];

    if (t < 196) {
        int c = t / 49, r = t % 49;
        int ky = r / 7, kx = r % 7;
        int iy = oy * 7 + ky, ix = ox * 7 + kx;
        float v;
        if (c < 3) v = img[(((b * 3 + c) * 448) + iy) * 448 + ix];
        else       v = msk[((b * 448) + iy) * 448 + ix];
        patch[t] = v;
    }
    __syncthreads();

    float acc = bias[t];
    const float* wr = w + t * 196;
    #pragma unroll 7
    for (int j = 0; j < 196; ++j) acc = fmaf(wr[j], patch[j], acc);

    // LayerNorm over the 256 channels of this pixel
    sb[t] = acc; __syncthreads();
    #pragma unroll
    for (int s = 128; s > 0; s >>= 1) { if (t < s) sb[t] += sb[t + s]; __syncthreads(); }
    float mean = sb[0] * (1.0f / 256.0f);
    __syncthreads();
    float d = acc - mean;
    sb[t] = d * d; __syncthreads();
    #pragma unroll
    for (int s = 128; s > 0; s >>= 1) { if (t < s) sb[t] += sb[t + s]; __syncthreads(); }
    float var = sb[0] * (1.0f / 256.0f);

    float xn = d * rsqrtf(var + 1e-6f);
    float y  = fmaf(lnw[t], xn, lnb[t]);
    y1[(pix << 8) + t] = (_Float16)gelu_exact(y);   // NHWC
}

// ---------------------------------------------------------------------------
// weight repacks -> f16 [N][K] (K contiguous per row => B-frag = 1 contig load)
// ---------------------------------------------------------------------------
__global__ void k_pack_w2(const float* __restrict__ w, _Float16* __restrict__ out) {
    // c2w [oc][ic][2][2] -> out[oc][ (ky*2+kx)*256 + ic ]
    int i = blockIdx.x * 256 + threadIdx.x;    // 1M
    int oc = i >> 10, k = i & 1023;
    int kc = k >> 8, ic = k & 255;
    int ky = kc >> 1, kx = kc & 1;
    out[i] = (_Float16)w[((oc * 256 + ic) * 2 + ky) * 2 + kx];
}

__global__ void k_pack_w3(const float* __restrict__ w, _Float16* __restrict__ out) {
    int i = blockIdx.x * 256 + threadIdx.x;    // 1M
    out[i] = (_Float16)w[i];
}

// ---------------------------------------------------------------------------
// WMMA fragment helpers (wave32, v_wmma_f32_16x16x32_f16, A/B in contiguous k order k = 16*(lane>>4) + s)
// ---------------------------------------------------------------------------
__device__ __forceinline__ v16h load_a_frag(const _Float16* __restrict__ row, int kh) {
    // lane<16 (kh=0): K 0..15 ; lane>=16 (kh=1): K 16..31
    v8h lo = *(const v8h*)(row + kh * 16);
    v8h hi = *(const v8h*)(row + kh * 16 + 8);
    v16h a;
    #pragma unroll
    for (int i = 0; i < 8; ++i) { a[i] = lo[i]; a[i + 8] = hi[i]; }
    return a;
}

__device__ __forceinline__ v16h load_b_frag(const _Float16* __restrict__ p) {
    // column n = lane&15 of B^T row-major [N][K]; K half = (lane>>4)*16, 16 contiguous halves
    return *(const v16h*)p;
}

// ---------------------------------------------------------------------------
// conv2 as GEMM: M=8192 (b,oy,ox), K=1024 (4 chunks of 256 ic), N=1024
// block = 8 waves (4M x 2N), wave tile 16x64, block tile 64x128
// ---------------------------------------------------------------------------
__global__ void k_gemm_conv2(const _Float16* __restrict__ y1,   // [B,64,64,256] NHWC
                             const _Float16* __restrict__ w2,   // [N=1024][K=1024]
                             const float* __restrict__ c2b,
                             float* __restrict__ x2) {           // [8192][1024]
    int lane = threadIdx.x & 31;
    int wave = threadIdx.x >> 5;
    int wm = wave >> 1, wn = wave & 1;
    int mBase = blockIdx.y * 64 + wm * 16;
    int nBase = blockIdx.x * 128 + wn * 64;

    int kh  = lane >> 4;
    int nlo = lane & 15;

    // decompose A row for this lane
    int am = mBase + nlo;          // M row for A-frag (lane&15)
    int b  = am >> 10;
    int r  = am & 1023;
    int oy = r >> 5, ox = r & 31;

    v8f acc[4];
    #pragma unroll
    for (int j = 0; j < 4; ++j) acc[j] = (v8f){0.f,0.f,0.f,0.f,0.f,0.f,0.f,0.f};

    #pragma unroll
    for (int kc = 0; kc < 4; ++kc) {           // (ky,kx) chunk, 256 ic each
        int ky = kc >> 1, kx = kc & 1;
        const _Float16* arow = y1 + (((b * 64 + oy * 2 + ky) * 64) + (ox * 2 + kx)) * 256;
        #pragma unroll
        for (int kk = 0; kk < 8; ++kk) {       // 32-wide K steps within chunk
            int k0 = kk * 32;
            v16h av = load_a_frag(arow + k0, kh);
            #pragma unroll
            for (int j = 0; j < 4; ++j) {
                const _Float16* bp = w2 + (size_t)(nBase + j * 16 + nlo) * 1024
                                        + kc * 256 + k0 + kh * 16;
                v16h bv = load_b_frag(bp);
                acc[j] = __builtin_amdgcn_wmma_f32_16x16x32_f16(
                             false, av, false, bv, (short)0, acc[j], false, false);
            }
        }
    }

    // epilogue: D layout — vgpr r, lane l: M = r + 8*(l>>4), N = l&15
    #pragma unroll
    for (int j = 0; j < 4; ++j) {
        int n = nBase + j * 16 + nlo;
        float bb = c2b[n];
        #pragma unroll
        for (int rr = 0; rr < 8; ++rr) {
            int m = mBase + rr + kh * 8;
            x2[(size_t)m * 1024 + n] = acc[j][rr] + bb;
        }
    }
}

// ---------------------------------------------------------------------------
// LN over 1024 channels per pixel + GELU -> f16 [8192][1024]
// ---------------------------------------------------------------------------
__global__ void k_ln2_gelu(const float* __restrict__ x2,
                           const float* __restrict__ lnw,
                           const float* __restrict__ lnb,
                           _Float16* __restrict__ y2) {
    int pix = blockIdx.x;          // 8192
    int t = threadIdx.x;           // 256
    const float* row = x2 + (size_t)pix * 1024;

    __shared__ float sb[256];
    float v[4]; float s = 0.f;
    #pragma unroll
    for (int i = 0; i < 4; ++i) { v[i] = row[t + i * 256]; s += v[i]; }
    sb[t] = s; __syncthreads();
    #pragma unroll
    for (int st = 128; st > 0; st >>= 1) { if (t < st) sb[t] += sb[t + st]; __syncthreads(); }
    float mean = sb[0] * (1.0f / 1024.0f);
    __syncthreads();
    float s2 = 0.f; float d[4];
    #pragma unroll
    for (int i = 0; i < 4; ++i) { d[i] = v[i] - mean; s2 += d[i] * d[i]; }
    sb[t] = s2; __syncthreads();
    #pragma unroll
    for (int st = 128; st > 0; st >>= 1) { if (t < st) sb[t] += sb[t + st]; __syncthreads(); }
    float rinv = rsqrtf(sb[0] * (1.0f / 1024.0f) + 1e-6f);

    #pragma unroll
    for (int i = 0; i < 4; ++i) {
        int c = t + i * 256;
        float y = fmaf(lnw[c], d[i] * rinv, lnb[c]);
        y2[(size_t)pix * 1024 + c] = (_Float16)gelu_exact(y);
    }
}

// ---------------------------------------------------------------------------
// conv3 1x1 as GEMM M=8192 K=1024 N=1024, fused + c3b + image_features add
// ---------------------------------------------------------------------------
__global__ void k_gemm_conv3(const _Float16* __restrict__ y2,   // [8192][1024]
                             const _Float16* __restrict__ w3,   // [N][K]
                             const float* __restrict__ c3b,
                             const float* __restrict__ feat,    // [8192][1024] (B,P,EMB == NHWC)
                             float* __restrict__ x3) {
    int lane = threadIdx.x & 31;
    int wave = threadIdx.x >> 5;
    int wm = wave >> 1, wn = wave & 1;
    int mBase = blockIdx.y * 64 + wm * 16;
    int nBase = blockIdx.x * 128 + wn * 64;
    int kh  = lane >> 4;
    int nlo = lane & 15;

    const _Float16* arow = y2 + (size_t)(mBase + nlo) * 1024;

    v8f acc[4];
    #pragma unroll
    for (int j = 0; j < 4; ++j) acc[j] = (v8f){0.f,0.f,0.f,0.f,0.f,0.f,0.f,0.f};

    #pragma unroll 8
    for (int ks = 0; ks < 32; ++ks) {
        int k0 = ks * 32;
        v16h av = load_a_frag(arow + k0, kh);
        #pragma unroll
        for (int j = 0; j < 4; ++j) {
            const _Float16* bp = w3 + (size_t)(nBase + j * 16 + nlo) * 1024 + k0 + kh * 16;
            v16h bv = load_b_frag(bp);
            acc[j] = __builtin_amdgcn_wmma_f32_16x16x32_f16(
                         false, av, false, bv, (short)0, acc[j], false, false);
        }
    }

    #pragma unroll
    for (int j = 0; j < 4; ++j) {
        int n = nBase + j * 16 + nlo;
        float bb = c3b[n];
        #pragma unroll
        for (int rr = 0; rr < 8; ++rr) {
            int m = mBase + rr + kh * 8;
            size_t idx = (size_t)m * 1024 + n;
            x3[idx] = acc[j][rr] + bb + feat[idx];
        }
    }
}

// ---------------------------------------------------------------------------
// bilinear point-sample + masked mean pool.  Validity == (x coord >= 0).
// grid: 8 batches x 4 channel chunks; thread = one channel
// ---------------------------------------------------------------------------
__global__ void k_pool(const float* __restrict__ x3,      // [B][32][32][1024]
                       const float* __restrict__ coords,  // [B][P][2]
                       float* __restrict__ pooled) {       // [B][1024]
    int b = blockIdx.x >> 2;
    int c = (blockIdx.x & 3) * 256 + threadIdx.x;
    const float* feat = x3 + (size_t)b * 1024 * 1024;
    const float2* pc = (const float2*)(coords + (size_t)b * NPTS * 2);

    float sum = 0.f, cnt = 0.f;
    for (int p = 0; p < NPTS; ++p) {
        float2 q = pc[p];
        if (q.x < 0.f) continue;            // invalid point (coord == -1)
        cnt += 1.f;
        float gx = q.x * 32.f - 0.5f;       // ((2u-1)+1)*0.5*32 - 0.5
        float gy = q.y * 32.f - 0.5f;
        float x0f = floorf(gx), y0f = floorf(gy);
        int x0 = (int)x0f, y0 = (int)y0f;
        float wx1 = gx - x0f, wx0 = 1.f - wx1;
        float wy1 = gy - y0f, wy0 = 1.f - wy1;
        float v00 = (x0   >= 0 && x0   < 32 && y0   >= 0 && y0   < 32) ? feat[(((y0)   * 32 + x0  ) << 10) + c] : 0.f;
        float v10 = (x0+1 >= 0 && x0+1 < 32 && y0   >= 0 && y0   < 32) ? feat[(((y0)   * 32 + x0+1) << 10) + c] : 0.f;
        float v01 = (x0   >= 0 && x0   < 32 && y0+1 >= 0 && y0+1 < 32) ? feat[(((y0+1) * 32 + x0  ) << 10) + c] : 0.f;
        float v11 = (x0+1 >= 0 && x0+1 < 32 && y0+1 >= 0 && y0+1 < 32) ? feat[(((y0+1) * 32 + x0+1) << 10) + c] : 0.f;
        sum += v00 * (wx0 * wy0) + v10 * (wx1 * wy0) + v01 * (wx0 * wy1) + v11 * (wx1 * wy1);
    }
    pooled[b * 1024 + c] = (cnt > 0.f) ? (sum / cnt) : 0.f;   // nan_to_num
}

// ---------------------------------------------------------------------------
// up-projection: out[b][o] = pooled[b] . up_w[o] + up_b[o]
// ---------------------------------------------------------------------------
__global__ void k_up(const float* __restrict__ pooled,
                     const float* __restrict__ w,     // [4096][1024]
                     const float* __restrict__ bias,  // [4096]
                     float* __restrict__ out) {        // [8][4096]
    int o = blockIdx.x * 256 + threadIdx.x;            // 4096
    float acc[BATCH];
    #pragma unroll
    for (int b = 0; b < BATCH; ++b) acc[b] = 0.f;
    const float* wr = w + (size_t)o * 1024;
    for (int k = 0; k < 1024; ++k) {
        float wv = wr[k];
        #pragma unroll
        for (int b = 0; b < BATCH; ++b) acc[b] = fmaf(pooled[b * 1024 + k], wv, acc[b]);
    }
    float bo = bias[o];
    #pragma unroll
    for (int b = 0; b < BATCH; ++b) out[b * 4096 + o] = acc[b] + bo;
}

// ---------------------------------------------------------------------------
extern "C" void kernel_launch(void* const* d_in, const int* in_sizes, int n_in,
                              void* d_out, int out_size, void* d_ws, size_t ws_size,
                              hipStream_t stream) {
    const float* images   = (const float*)d_in[0];
    const float* masks    = (const float*)d_in[1];
    const float* feats    = (const float*)d_in[2];   // [8,1024,1024]
    const float* coords   = (const float*)d_in[3];   // [8,P,2]
    // d_in[4] valid_mask (bool) unused: validity derived from coords >= 0
    const float* c1w = (const float*)d_in[5];
    const float* c1b = (const float*)d_in[6];
    const float* ln1w = (const float*)d_in[7];
    const float* ln1b = (const float*)d_in[8];
    const float* c2w = (const float*)d_in[9];
    const float* c2b = (const float*)d_in[10];
    const float* ln2w = (const float*)d_in[11];
    const float* ln2b = (const float*)d_in[12];
    const float* c3w = (const float*)d_in[13];
    const float* c3b = (const float*)d_in[14];
    const float* up_w = (const float*)d_in[15];
    const float* up_b = (const float*)d_in[16];

    char* ws = (char*)d_ws;
    _Float16* y1h    = (_Float16*)(ws);                        // 16.78 MB
    _Float16* w2h    = (_Float16*)(ws + 16777216);             //  2.10 MB
    _Float16* w3h    = (_Float16*)(ws + 18874368);             //  2.10 MB
    float*    x2     = (float*)   (ws + 20971520);             // 33.55 MB
    _Float16* y2h    = (_Float16*)(ws + 54525952);             // 16.78 MB
    float*    pooled = (float*)   (ws + 71303168);             // 32 KB
    float*    x3     = x2;   // x2 consumed by LN2 before conv3 writes

    k_pack_w2<<<4096, 256, 0, stream>>>(c2w, w2h);
    k_pack_w3<<<4096, 256, 0, stream>>>(c3w, w3h);
    k_conv1_ln_gelu<<<32768, 256, 0, stream>>>(images, masks, c1w, c1b, ln1w, ln1b, y1h);
    k_gemm_conv2<<<dim3(8, 128), 256, 0, stream>>>(y1h, w2h, c2b, x2);
    k_ln2_gelu<<<8192, 256, 0, stream>>>(x2, ln2w, ln2b, y2h);
    k_gemm_conv3<<<dim3(8, 128), 256, 0, stream>>>(y2h, w3h, c3b, feats, x3);
    k_pool<<<32, 256, 0, stream>>>(x3, coords, pooled);
    k_up<<<16, 256, 0, stream>>>(pooled, up_w, up_b, (float*)d_out);
}
